// MixGraphEncoder_57123065037605
// MI455X (gfx1250) — compile-verified, matrix operand order checked by reading, not run on hardware
//
#include <hip/hip_runtime.h>
#include <hip/hip_bf16.h>

#define DEV __device__ __forceinline__

typedef __attribute__((ext_vector_type(16))) __bf16 v16bf;
typedef __attribute__((ext_vector_type(8)))  float  v8f;

union FragU { v16bf v; uint4 q[2]; unsigned int u[8]; };

constexpr int Hh   = 256;
constexpr int EHh  = 64;
constexpr int EINh = 16;
constexpr int Bg   = 65536;
constexpr int Nn   = 3 * Bg;
constexpr int Ee   = 3 * Bg;
constexpr int Lnum = 2;

DEV float gelu_f(float x) { return 0.5f * x * (1.0f + erff(x * 0.70710678118654752440f)); }

DEV v8f wmma_bf16(v16bf a, v16bf b, v8f c) {
  return __builtin_amdgcn_wmma_f32_16x16x32_bf16(false, a, false, b, (short)0, c, false, false);
}

// A fragment (16x32 bf16) from a bf16 LDS tile, stride in elements (multiple of 8).
// ISA 7.12.2 16-bit A layout: v0-3 <-> K k0+{0..7} (lanes0-15) / k0+{8..15} (lanes16-31),
// v4-7 <-> +16. Per lane: two contiguous 16B LDS reads.
DEV v16bf load_a_lds(const __bf16* lds, int stride, int k0) {
  int l = threadIdx.x & 31;
  const __bf16* r = lds + (l & 15) * stride + k0 + ((l >> 4) << 3);
  FragU f;
  f.q[0] = *(const uint4*)(r);
  f.q[1] = *(const uint4*)(r + 16);
  return f.v;
}
// B fragment from pre-packed weights: contiguous 32B per lane per tile.
DEV v16bf load_b_packed(const __bf16* P, int CT, int kt, int ct) {
  int l = threadIdx.x & 31;
  const uint4* s = (const uint4*)(P + ((((kt * CT + ct) << 5) + l) << 4));
  FragU f;
  f.q[0] = s[0];
  f.q[1] = s[1];
  return f.v;
}

DEV float rsum8(float v) {
  v += __shfl_down(v, 4, 8);
  v += __shfl_down(v, 2, 8);
  v += __shfl_down(v, 1, 8);
  return __shfl(v, 0, 8);
}
DEV float rsum32(float v) {
#pragma unroll
  for (int o = 16; o > 0; o >>= 1) v += __shfl_down(v, o, 32);
  return __shfl(v, 0, 32);
}

// ---------------- weight packing: f32 (K x N) -> fragment-ordered bf16 tiles
__global__ void pack_weight(const float* __restrict__ W, __bf16* __restrict__ P,
                            int K, int Ncols) {
  int idx = blockIdx.x * blockDim.x + threadIdx.x;
  if (idx >= K * Ncols) return;
  int e = idx & 15;
  int l = (idx >> 4) & 31;
  int t = idx >> 9;
  int CT = Ncols >> 4;
  int ct = t % CT, kt = t / CT;
  int v = e >> 1, pbit = e & 1;
  int k = kt * 32 + ((v < 4) ? 0 : 16) + ((l >= 16) ? 8 : 0) + 2 * (v & 3) + pbit;
  int n = ct * 16 + (l & 15);
  P[idx] = (__bf16)W[k * Ncols + n];
}

// ---------------- e = LN(gelu(edge_attr @ ein_w + ein_b))
__global__ void edge_init_kernel(const float* __restrict__ ea,
                                 const float* __restrict__ w, const float* __restrict__ b,
                                 const float* __restrict__ g, const float* __restrict__ bb,
                                 float* __restrict__ e) {
  int row = blockIdx.x * 8 + (threadIdx.x >> 5);
  int lane = threadIdx.x & 31;
  const float* ar = ea + (size_t)row * EINh;
  int c0 = lane, c1 = lane + 32;
  float a0 = b[c0], a1 = b[c1];
#pragma unroll
  for (int k = 0; k < EINh; ++k) {
    float av = ar[k];
    a0 += av * w[k * EHh + c0];
    a1 += av * w[k * EHh + c1];
  }
  a0 = gelu_f(a0); a1 = gelu_f(a1);
  float mu = rsum32(a0 + a1) * (1.0f / EHh);
  float d0 = a0 - mu, d1 = a1 - mu;
  float rstd = rsqrtf(rsum32(d0 * d0 + d1 * d1) * (1.0f / EHh) + 1e-5f);
  float* er = e + (size_t)row * EHh;
  er[c0] = d0 * rstd * g[c0] + bb[c0];
  er[c1] = d1 * rstd * g[c1] + bb[c1];
}

// ---------------- edge update: gate + 2-layer MLP + residual + LN (in-place e)
__global__ __launch_bounds__(128) void edge_update_kernel(
    const float* __restrict__ h, float* __restrict__ e,
    const int* __restrict__ src, const int* __restrict__ dst,
    const __bf16* __restrict__ w1p, const float* __restrict__ b1,
    const __bf16* __restrict__ w2p, const float* __restrict__ b2,
    const float* __restrict__ gw, const float* __restrict__ gb,
    const float* __restrict__ lng, const float* __restrict__ lnb) {
  constexpr int ZW = 2 * Hh + EHh, ZS = ZW + 8;   // 576 / 584 (stride mult of 8)
  __shared__ __bf16 z[16 * ZS];
  __shared__ float gate[16];
  __shared__ __bf16 a1[16 * EHh];
  __shared__ float buf[16 * EHh];
  __shared__ int sidx[16], didx[16];
  int tid = threadIdx.x;
  int e0 = blockIdx.x * 16;
  if (tid < 16) { sidx[tid] = src[e0 + tid]; didx[tid] = dst[e0 + tid]; }
  __syncthreads();
  for (int idx = tid; idx < 16 * ZW; idx += 128) {
    int row = idx / ZW, c = idx - row * ZW;
    float v;
    if (c < Hh)            v = h[(size_t)sidx[row] * Hh + c];
    else if (c < 2 * Hh)   v = h[(size_t)didx[row] * Hh + (c - Hh)];
    else                   v = e[(size_t)(e0 + row) * EHh + (c - 2 * Hh)];
    z[row * ZS + c] = (__bf16)v;
  }
  __syncthreads();
  { // sigmoid gate: z . egate_w + egate_b
    int row = tid >> 3, sub = tid & 7;
    float s = 0.f;
    for (int c = sub; c < ZW; c += 8) s += (float)z[row * ZS + c] * gw[c];
    s = rsum8(s);
    if (sub == 0) gate[row] = 1.0f / (1.0f + __expf(-(s + gb[0])));
  }
  int wave = tid >> 5, lane = tid & 31;
  { // GEMM1: 576 -> 64, gelu
    int ct = wave;
    v8f acc = {0, 0, 0, 0, 0, 0, 0, 0};
    for (int kt = 0; kt < 18; ++kt) {
      if (kt < 17) __builtin_prefetch(w1p + (((kt + 1) * 4 + ct) << 9), 0, 1);
      v16bf a = load_a_lds(z, ZS, kt * 32);
      v16bf bf = load_b_packed(w1p, 4, kt, ct);
      acc = wmma_bf16(a, bf, acc);
    }
    int n = ct * 16 + (lane & 15);
    int mb = (lane >= 16) ? 8 : 0;
    float bias = b1[n];
#pragma unroll
    for (int v = 0; v < 8; ++v)
      a1[(mb + v) * EHh + n] = (__bf16)gelu_f(acc[v] + bias);
  }
  __syncthreads();
  { // GEMM2: 64 -> 64, gated scaled residual (e_old re-read from global)
    int ct = wave;
    v8f acc = {0, 0, 0, 0, 0, 0, 0, 0};
    for (int kt = 0; kt < 2; ++kt) {
      v16bf a = load_a_lds(a1, EHh, kt * 32);
      v16bf bf = load_b_packed(w2p, 4, kt, ct);
      acc = wmma_bf16(a, bf, acc);
    }
    int n = ct * 16 + (lane & 15);
    int mb = (lane >= 16) ? 8 : 0;
    float bias = b2[n];
#pragma unroll
    for (int v = 0; v < 8; ++v) {
      int M = mb + v;
      float eold = e[(size_t)(e0 + M) * EHh + n];
      buf[M * EHh + n] = eold + 0.1f * gate[M] * (acc[v] + bias);
    }
  }
  __syncthreads();
  { // LN over 64 -> e
    int row = tid >> 3, sub = tid & 7;
    float s = 0.f;
    for (int n = sub; n < EHh; n += 8) s += buf[row * EHh + n];
    float mu = rsum8(s) * (1.0f / EHh);
    float q = 0.f;
    for (int n = sub; n < EHh; n += 8) { float d = buf[row * EHh + n] - mu; q += d * d; }
    float rstd = rsqrtf(rsum8(q) * (1.0f / EHh) + 1e-5f);
    float* er = e + (size_t)(e0 + row) * EHh;
    for (int n = sub; n < EHh; n += 8)
      er[n] = (buf[row * EHh + n] - mu) * rstd * lng[n] + lnb[n];
  }
}

// ---------------- message: gelu([hs,e]@W1+b1)@W2+b2, scatter-add to agg[dst]
__global__ __launch_bounds__(128) void message_kernel(
    const float* __restrict__ h, const float* __restrict__ e,
    const int* __restrict__ src, const int* __restrict__ dst,
    const __bf16* __restrict__ w1p, const float* __restrict__ b1,
    const __bf16* __restrict__ w2p, const float* __restrict__ b2,
    float* __restrict__ agg) {
  constexpr int ZW = Hh + EHh, ZS = ZW + 8;   // 320 / 328
  __shared__ __bf16 z[16 * ZS];
  __shared__ __bf16 m1[16 * Hh];
  __shared__ int sidx[16], didx[16];
  int tid = threadIdx.x, e0 = blockIdx.x * 16;
  if (tid < 16) { sidx[tid] = src[e0 + tid]; didx[tid] = dst[e0 + tid]; }
  __syncthreads();
  for (int idx = tid; idx < 16 * ZW; idx += 128) {
    int row = idx / ZW, c = idx - row * ZW;
    float v = (c < Hh) ? h[(size_t)sidx[row] * Hh + c]
                       : e[(size_t)(e0 + row) * EHh + (c - Hh)];
    z[row * ZS + c] = (__bf16)v;
  }
  __syncthreads();
  int wave = tid >> 5, lane = tid & 31;
  int n0c = (lane & 15);
  int mb = (lane >= 16) ? 8 : 0;
  { // GEMM1: 320 -> 256, gelu; K-outer, 4 column tiles per wave
    v8f acc[4] = {{0,0,0,0,0,0,0,0},{0,0,0,0,0,0,0,0},{0,0,0,0,0,0,0,0},{0,0,0,0,0,0,0,0}};
    for (int kt = 0; kt < 10; ++kt) {
      if (kt < 9) __builtin_prefetch(w1p + ((((kt + 1) * 16 + wave) << 5) << 4), 0, 1);
      v16bf a = load_a_lds(z, ZS, kt * 32);
#pragma unroll
      for (int cc = 0; cc < 4; ++cc) {
        v16bf bf = load_b_packed(w1p, 16, kt, wave + 4 * cc);
        acc[cc] = wmma_bf16(a, bf, acc[cc]);
      }
    }
#pragma unroll
    for (int cc = 0; cc < 4; ++cc) {
      int n = (wave + 4 * cc) * 16 + n0c;
      float bias = b1[n];
#pragma unroll
      for (int v = 0; v < 8; ++v)
        m1[(mb + v) * Hh + n] = (__bf16)gelu_f(acc[cc][v] + bias);
    }
  }
  __syncthreads();
  { // GEMM2: 256 -> 256, atomic scatter to agg[dst]
    v8f acc[4] = {{0,0,0,0,0,0,0,0},{0,0,0,0,0,0,0,0},{0,0,0,0,0,0,0,0},{0,0,0,0,0,0,0,0}};
    for (int kt = 0; kt < 8; ++kt) {
      if (kt < 7) __builtin_prefetch(w2p + ((((kt + 1) * 16 + wave) << 5) << 4), 0, 1);
      v16bf a = load_a_lds(m1, Hh, kt * 32);
#pragma unroll
      for (int cc = 0; cc < 4; ++cc) {
        v16bf bf = load_b_packed(w2p, 16, kt, wave + 4 * cc);
        acc[cc] = wmma_bf16(a, bf, acc[cc]);
      }
    }
#pragma unroll
    for (int cc = 0; cc < 4; ++cc) {
      int n = (wave + 4 * cc) * 16 + n0c;
      float bias = b2[n];
#pragma unroll
      for (int v = 0; v < 8; ++v) {
        int M = mb + v;
        unsafeAtomicAdd(&agg[(size_t)didx[M] * Hh + n], acc[cc][v] + bias);
      }
    }
  }
}

// ---------------- node update: gelu([h,agg]@W1+b1)@W2+b2, +h, LN (in-place h)
__global__ __launch_bounds__(128) void node_update_kernel(
    float* __restrict__ h, const float* __restrict__ agg,
    const __bf16* __restrict__ w1p, const float* __restrict__ b1,
    const __bf16* __restrict__ w2p, const float* __restrict__ b2,
    const float* __restrict__ lng, const float* __restrict__ lnb) {
  constexpr int ZW = 2 * Hh, ZS = ZW + 8;   // 512 / 520
  constexpr int LNS = Hh + 4;               // 260
  __shared__ __bf16 z[16 * ZS];
  __shared__ __bf16 m1[16 * Hh];
  __shared__ float lnbuf[16 * LNS];
  int tid = threadIdx.x, n0 = blockIdx.x * 16;
  for (int idx = tid; idx < 16 * ZW; idx += 128) {
    int row = idx / ZW, c = idx - row * ZW;
    float v = (c < Hh) ? h[(size_t)(n0 + row) * Hh + c]
                       : agg[(size_t)(n0 + row) * Hh + (c - Hh)];
    z[row * ZS + c] = (__bf16)v;
  }
  __syncthreads();
  int wave = tid >> 5, lane = tid & 31;
  int n0c = (lane & 15);
  int mb = (lane >= 16) ? 8 : 0;
  { // GEMM1: 512 -> 256, gelu
    v8f acc[4] = {{0,0,0,0,0,0,0,0},{0,0,0,0,0,0,0,0},{0,0,0,0,0,0,0,0},{0,0,0,0,0,0,0,0}};
    for (int kt = 0; kt < 16; ++kt) {
      if (kt < 15) __builtin_prefetch(w1p + ((((kt + 1) * 16 + wave) << 5) << 4), 0, 1);
      v16bf a = load_a_lds(z, ZS, kt * 32);
#pragma unroll
      for (int cc = 0; cc < 4; ++cc) {
        v16bf bf = load_b_packed(w1p, 16, kt, wave + 4 * cc);
        acc[cc] = wmma_bf16(a, bf, acc[cc]);
      }
    }
#pragma unroll
    for (int cc = 0; cc < 4; ++cc) {
      int n = (wave + 4 * cc) * 16 + n0c;
      float bias = b1[n];
#pragma unroll
      for (int v = 0; v < 8; ++v)
        m1[(mb + v) * Hh + n] = (__bf16)gelu_f(acc[cc][v] + bias);
    }
  }
  __syncthreads();
  { // GEMM2: 256 -> 256, residual (h re-read from global) into lnbuf
    v8f acc[4] = {{0,0,0,0,0,0,0,0},{0,0,0,0,0,0,0,0},{0,0,0,0,0,0,0,0},{0,0,0,0,0,0,0,0}};
    for (int kt = 0; kt < 8; ++kt) {
      if (kt < 7) __builtin_prefetch(w2p + ((((kt + 1) * 16 + wave) << 5) << 4), 0, 1);
      v16bf a = load_a_lds(m1, Hh, kt * 32);
#pragma unroll
      for (int cc = 0; cc < 4; ++cc) {
        v16bf bf = load_b_packed(w2p, 16, kt, wave + 4 * cc);
        acc[cc] = wmma_bf16(a, bf, acc[cc]);
      }
    }
#pragma unroll
    for (int cc = 0; cc < 4; ++cc) {
      int n = (wave + 4 * cc) * 16 + n0c;
      float bias = b2[n];
#pragma unroll
      for (int v = 0; v < 8; ++v) {
        int M = mb + v;
        float hold = h[(size_t)(n0 + M) * Hh + n];
        lnbuf[M * LNS + n] = hold + acc[cc][v] + bias;
      }
    }
  }
  __syncthreads();
  { // LN over 256 -> h
    int row = tid >> 3, sub = tid & 7;
    float s = 0.f;
    for (int n = sub; n < Hh; n += 8) s += lnbuf[row * LNS + n];
    float mu = rsum8(s) * (1.0f / Hh);
    float q = 0.f;
    for (int n = sub; n < Hh; n += 8) { float d = lnbuf[row * LNS + n] - mu; q += d * d; }
    float rstd = rsqrtf(rsum8(q) * (1.0f / Hh) + 1e-5f);
    float* hr = h + (size_t)(n0 + row) * Hh;
    for (int n = sub; n < Hh; n += 8)
      hr[n] = (lnbuf[row * LNS + n] - mu) * rstd * lng[n] + lnb[n];
  }
}

// ---------------- node_out = LN(h); atomic pool into mix; count per graph
__global__ void final_kernel(const float* __restrict__ h, const int* __restrict__ batch,
                             const float* __restrict__ g, const float* __restrict__ b,
                             float* __restrict__ out, float* __restrict__ mix,
                             float* __restrict__ cnt) {
  int row = blockIdx.x * 8 + (threadIdx.x >> 5);
  int lane = threadIdx.x & 31;
  const float* hr = h + (size_t)row * Hh;
  float x[8];
  float s = 0.f;
#pragma unroll
  for (int j = 0; j < 8; ++j) { x[j] = hr[lane + 32 * j]; s += x[j]; }
  float mu = rsum32(s) * (1.0f / Hh);
  float q = 0.f;
#pragma unroll
  for (int j = 0; j < 8; ++j) { float d = x[j] - mu; q += d * d; }
  float rstd = rsqrtf(rsum32(q) * (1.0f / Hh) + 1e-5f);
  int gr = batch[row];
  float* mr = mix + (size_t)gr * Hh;
  float* orow = out + (size_t)row * Hh;
#pragma unroll
  for (int j = 0; j < 8; ++j) {
    int c = lane + 32 * j;
    float v = (x[j] - mu) * rstd * g[c] + b[c];
    orow[c] = v;
    unsafeAtomicAdd(&mr[c], v);
  }
  if (lane == 0) unsafeAtomicAdd(&cnt[gr], 1.0f);
}

__global__ void mix_div_kernel(float* __restrict__ mix, const float* __restrict__ cnt) {
  int idx = blockIdx.x * blockDim.x + threadIdx.x;
  if (idx < Bg * Hh) mix[idx] /= fmaxf(cnt[idx >> 8], 1.0f);
}

extern "C" void kernel_launch(void* const* d_in, const int* in_sizes, int n_in,
                              void* d_out, int out_size, void* d_ws, size_t ws_size,
                              hipStream_t stream) {
  (void)in_sizes; (void)n_in; (void)out_size; (void)ws_size;
  const float* node_h    = (const float*)d_in[0];
  const float* edge_attr = (const float*)d_in[1];
  const int*   eidx      = (const int*)d_in[2];
  const int*   batch     = (const int*)d_in[3];
  const float* ein_w  = (const float*)d_in[5];
  const float* ein_b  = (const float*)d_in[6];
  const float* enorm_g = (const float*)d_in[7];
  const float* enorm_b = (const float*)d_in[8];
  const float* msg_w1 = (const float*)d_in[9];
  const float* msg_b1 = (const float*)d_in[10];
  const float* msg_w2 = (const float*)d_in[11];
  const float* msg_b2 = (const float*)d_in[12];
  const float* upd_w1 = (const float*)d_in[13];
  const float* upd_b1 = (const float*)d_in[14];
  const float* upd_w2 = (const float*)d_in[15];
  const float* upd_b2 = (const float*)d_in[16];
  const float* nrm_g  = (const float*)d_in[17];
  const float* nrm_b  = (const float*)d_in[18];
  const float* emlp_w1 = (const float*)d_in[19];
  const float* emlp_b1 = (const float*)d_in[20];
  const float* emlp_w2 = (const float*)d_in[21];
  const float* emlp_b2 = (const float*)d_in[22];
  const float* egate_w = (const float*)d_in[23];
  const float* egate_b = (const float*)d_in[24];
  const float* eln_g   = (const float*)d_in[25];
  const float* eln_b   = (const float*)d_in[26];
  const float* onorm_g = (const float*)d_in[27];
  const float* onorm_b = (const float*)d_in[28];

  const int* src = eidx;
  const int* dst = eidx + Ee;

  char* p = (char*)d_ws;
  float* h   = (float*)p;  p += (size_t)Nn * Hh * 4;
  float* agg = (float*)p;  p += (size_t)Nn * Hh * 4;
  float* e   = (float*)p;  p += (size_t)Ee * EHh * 4;
  float* cnt = (float*)p;  p += (size_t)Bg * 4;
  __bf16* msg_w1p  = (__bf16*)p; p += (size_t)Lnum * 320 * 256 * 2;
  __bf16* msg_w2p  = (__bf16*)p; p += (size_t)Lnum * 256 * 256 * 2;
  __bf16* upd_w1p  = (__bf16*)p; p += (size_t)Lnum * 512 * 256 * 2;
  __bf16* upd_w2p  = (__bf16*)p; p += (size_t)Lnum * 256 * 256 * 2;
  __bf16* emlp_w1p = (__bf16*)p; p += (size_t)Lnum * 576 * 64 * 2;
  __bf16* emlp_w2p = (__bf16*)p; p += (size_t)Lnum * 64 * 64 * 2;

  float* out = (float*)d_out;
  float* mix = out + (size_t)Nn * Hh;

  for (int i = 0; i < Lnum; ++i) {
    pack_weight<<<(320 * 256 + 255) / 256, 256, 0, stream>>>(msg_w1 + i * 320 * 256, msg_w1p + i * 320 * 256, 320, 256);
    pack_weight<<<(256 * 256 + 255) / 256, 256, 0, stream>>>(msg_w2 + i * 256 * 256, msg_w2p + i * 256 * 256, 256, 256);
    pack_weight<<<(512 * 256 + 255) / 256, 256, 0, stream>>>(upd_w1 + i * 512 * 256, upd_w1p + i * 512 * 256, 512, 256);
    pack_weight<<<(256 * 256 + 255) / 256, 256, 0, stream>>>(upd_w2 + i * 256 * 256, upd_w2p + i * 256 * 256, 256, 256);
    pack_weight<<<(576 * 64 + 255) / 256, 256, 0, stream>>>(emlp_w1 + i * 576 * 64, emlp_w1p + i * 576 * 64, 576, 64);
    pack_weight<<<(64 * 64 + 255) / 256, 256, 0, stream>>>(emlp_w2 + i * 64 * 64, emlp_w2p + i * 64 * 64, 64, 64);
  }

  hipMemcpyAsync(h, node_h, (size_t)Nn * Hh * 4, hipMemcpyDeviceToDevice, stream);
  hipMemsetAsync(cnt, 0, (size_t)Bg * 4, stream);
  hipMemsetAsync(mix, 0, (size_t)Bg * Hh * 4, stream);

  edge_init_kernel<<<Ee / 8, 256, 0, stream>>>(edge_attr, ein_w, ein_b, enorm_g, enorm_b, e);

  for (int i = 0; i < Lnum; ++i) {
    edge_update_kernel<<<Ee / 16, 128, 0, stream>>>(
        h, e, src, dst,
        emlp_w1p + i * 576 * 64, emlp_b1 + i * 64,
        emlp_w2p + i * 64 * 64,  emlp_b2 + i * 64,
        egate_w + i * 576, egate_b + i,
        eln_g + i * 64, eln_b + i * 64);
    hipMemsetAsync(agg, 0, (size_t)Nn * Hh * 4, stream);
    message_kernel<<<Ee / 16, 128, 0, stream>>>(
        h, e, src, dst,
        msg_w1p + i * 320 * 256, msg_b1 + i * 256,
        msg_w2p + i * 256 * 256, msg_b2 + i * 256, agg);
    node_update_kernel<<<Nn / 16, 128, 0, stream>>>(
        h, agg,
        upd_w1p + i * 512 * 256, upd_b1 + i * 256,
        upd_w2p + i * 256 * 256, upd_b2 + i * 256,
        nrm_g + i * 256, nrm_b + i * 256);
  }

  final_kernel<<<Nn / 8, 256, 0, stream>>>(h, batch, onorm_g, onorm_b, out, mix, cnt);
  mix_div_kernel<<<(Bg * Hh + 255) / 256, 256, 0, stream>>>(mix, cnt);
}
